// Qwen3NextGatedDeltaNet_54357106098372
// MI455X (gfx1250) — matrix operand order repaired and checked by reading, not compiled, not run on hardware
//
#include <hip/hip_runtime.h>
#include <hip/hip_bf16.h>
#include <cstddef>
#include <cstdint>

// ---------------------------------------------------------------------------
// Sizes (fixed by the reference)
// ---------------------------------------------------------------------------
#define TT      1024
#define HDIM    2048
#define HK_     16
#define HV_     32
#define DK_     128
#define DV_     128
#define KDIM_   (HK_ * DK_)              // 2048
#define VDIM_   (HV_ * DV_)              // 4096
#define QKVZ_N  (2 * KDIM_ + 2 * VDIM_)  // 12288
#define CONVD   (2 * KDIM_ + VDIM_)      // 8192
#define HEADSTRIDE (2 * DK_ + 2 * 2 * DV_) // 768 per HK head in qkvz

typedef __attribute__((ext_vector_type(16))) __bf16 v16bf;
typedef __attribute__((ext_vector_type(8)))  float  v8f;
typedef int v4i_t __attribute__((vector_size(16)));

// ---------------------------------------------------------------------------
// Async global->LDS copy (CDNA5 GLOBAL_LOAD_ASYNC_TO_LDS_B128, ASYNCcnt).
// Builtin confirmed present by the round-2 diagnostic; it takes
// (AS1 v4i*, AS3 v4i*, imm offset, imm cpol).
// ---------------------------------------------------------------------------
#if defined(__has_builtin)
#  if __has_builtin(__builtin_amdgcn_global_load_async_to_lds_b128) && \
      __has_builtin(__builtin_amdgcn_s_wait_asynccnt)
#    define USE_ASYNC 1
#  else
#    define USE_ASYNC 0
#  endif
#else
#  define USE_ASYNC 0
#endif

__device__ __forceinline__ void async_cp16(const void* g, void* l)
{
#if USE_ASYNC
    __builtin_amdgcn_global_load_async_to_lds_b128(
        (__attribute__((address_space(1))) v4i_t*)g,
        (__attribute__((address_space(3))) v4i_t*)l, 0, 0);
#else
    *(uint4*)l = *(const uint4*)g;
#endif
}

__device__ __forceinline__ void async_wait_all()
{
#if USE_ASYNC
    __builtin_amdgcn_s_wait_asynccnt(0);
#endif
}

__device__ __forceinline__ float sigmoidf_(float x) { return 1.0f / (1.0f + __expf(-x)); }
__device__ __forceinline__ float siluf_(float x)    { return x * sigmoidf_(x); }

// ---------------------------------------------------------------------------
// f32 -> bf16 elementwise convert (for A operands, row-major kept)
// ---------------------------------------------------------------------------
__global__ __launch_bounds__(256) void cvt_bf16_kernel(
    const float* __restrict__ in, __bf16* __restrict__ out, int n)
{
    int idx = blockIdx.x * 256 + threadIdx.x;
    if (idx < n) out[idx] = (__bf16)in[idx];
}

// ---------------------------------------------------------------------------
// W[K,N] f32 -> Wt[N,K] bf16 (tiled 32x32 transpose through LDS)
// ---------------------------------------------------------------------------
__global__ __launch_bounds__(256) void transpose_cvt_bf16_kernel(
    const float* __restrict__ W, __bf16* __restrict__ Wt, int K, int N)
{
    __shared__ float tile[32][33];
    const int bk = blockIdx.y * 32;
    const int bn = blockIdx.x * 32;
    const int tx = threadIdx.x & 31;
    const int ty = threadIdx.x >> 5;      // 8 rows per pass
#pragma unroll
    for (int i = 0; i < 32; i += 8)
        tile[ty + i][tx] = W[(size_t)(bk + ty + i) * N + bn + tx];
    __syncthreads();
#pragma unroll
    for (int i = 0; i < 32; i += 8)
        Wt[(size_t)(bn + ty + i) * K + bk + tx] = (__bf16)tile[tx][ty + i];
}

// ---------------------------------------------------------------------------
// bf16 WMMA GEMM: C[M,N] = A[M,K] @ Bt[N,K]^T, f32 accumulate.
// Block = 256 threads (8 waves), WG tile 128x64, wave tile 32x32 (2x2 wmma),
// K-step 32, double-buffered LDS staged with async global->LDS copies.
// Requires: M % 128 == 0, N % 64 == 0, K % 32 == 0.
// ---------------------------------------------------------------------------
__global__ __launch_bounds__(256) void wmma_gemm_bf16(
    const __bf16* __restrict__ A, const __bf16* __restrict__ Bt,
    float* __restrict__ C, int M, int N, int K)
{
    __shared__ __bf16 As[2][128 * 32];
    __shared__ __bf16 Bs[2][64 * 32];

    const int tid  = threadIdx.x;
    const int lane = tid & 31;
    const int w    = tid >> 5;
    const int wm   = w & 3;                 // 4 waves along M
    const int wn   = w >> 2;                // 2 waves along N
    const int tileM = blockIdx.y * 128;
    const int tileN = blockIdx.x * 64;

    // staging assignment: 16B chunks along K
    const int am  = tid >> 2;               // 0..63 (rows am, am+64)
    const int akq = (tid & 3) * 8;          // K sub-offset 0,8,16,24

    auto issue = [&](int buf, int kk) {
        async_cp16(A  + (size_t)(tileM + am)      * K + kk + akq, &As[buf][ am       * 32 + akq]);
        async_cp16(A  + (size_t)(tileM + am + 64) * K + kk + akq, &As[buf][(am + 64) * 32 + akq]);
        async_cp16(Bt + (size_t)(tileN + am)      * K + kk + akq, &Bs[buf][ am       * 32 + akq]);
    };

    v8f acc[2][2] = {};
    int buf = 0;
    issue(0, 0);

    for (int kk = 0; kk < K; kk += 32) {
        async_wait_all();
        __syncthreads();
        if (kk + 32 < K) issue(buf ^ 1, kk + 32);
        if (kk + 64 < K) {
            __builtin_prefetch(A  + (size_t)(tileM + am) * K + kk + 64, 0, 1);
            __builtin_prefetch(Bt + (size_t)(tileN + am) * K + kk + 64, 0, 1);
        }

        // ---- fragments (ISA 7.12.2 layouts) ----
        const int mrow  = lane & 15;
        const int khalf = lane >> 4;        // A: +8 K ; B: +8 K

        v16bf afr[2], bfr[2];
#pragma unroll
        for (int mi = 0; mi < 2; ++mi) {
            const int mbase = (wm * 32 + mi * 16 + mrow) * 32;
#pragma unroll
            for (int e = 0; e < 8; ++e) {
                afr[mi][e]     = As[buf][mbase + khalf * 8 + e];
                afr[mi][8 + e] = As[buf][mbase + 16 + khalf * 8 + e];
            }
        }
#pragma unroll
        for (int ni = 0; ni < 2; ++ni) {
            const int nbase = (wn * 32 + ni * 16 + mrow) * 32;
#pragma unroll
            for (int e = 0; e < 8; ++e) {
                bfr[ni][e]     = Bs[buf][nbase + khalf * 8 + e];
                bfr[ni][8 + e] = Bs[buf][nbase + 16 + khalf * 8 + e];
            }
        }

#pragma unroll
        for (int mi = 0; mi < 2; ++mi)
#pragma unroll
            for (int ni = 0; ni < 2; ++ni)
                acc[mi][ni] = __builtin_amdgcn_wmma_f32_16x16x32_bf16(
                    false, afr[mi], false, bfr[ni], (short)0, acc[mi][ni], false, false);

        buf ^= 1;
    }

    // ---- store D (lane: n = lane&15, M half = lane>>4) ----
    const int n  = lane & 15;
    const int mh = lane >> 4;
#pragma unroll
    for (int mi = 0; mi < 2; ++mi)
#pragma unroll
        for (int ni = 0; ni < 2; ++ni) {
            const int col = tileN + wn * 32 + ni * 16 + n;
#pragma unroll
            for (int v = 0; v < 8; ++v) {
                const int row = tileM + wm * 32 + mi * 16 + v + 8 * mh;
                C[(size_t)row * N + col] = acc[mi][ni][v];
            }
        }
}

// ---------------------------------------------------------------------------
// Depthwise causal conv (KSIZE=4) over permuted qkvz columns + SiLU.
// ---------------------------------------------------------------------------
__device__ __forceinline__ int conv_col(int c)
{
    if (c < KDIM_)      { int hh = c >> 7, d = c & 127;  return hh * HEADSTRIDE + d; }
    if (c < 2 * KDIM_)  { int c2 = c - KDIM_; int hh = c2 >> 7, d = c2 & 127;
                          return hh * HEADSTRIDE + DK_ + d; }
    int c3 = c - 2 * KDIM_; int hv = c3 >> 7, d = c3 & 127;
    return (hv >> 1) * HEADSTRIDE + 2 * DK_ + (hv & 1) * DV_ + d;
}

__global__ __launch_bounds__(256) void conv_silu_kernel(
    const float* __restrict__ qkvz, const float* __restrict__ conv_w,
    float* __restrict__ y)
{
    int idx = blockIdx.x * 256 + threadIdx.x;
    if (idx >= TT * CONVD) return;
    int t = idx >> 13;
    int c = idx & (CONVD - 1);
    int col = conv_col(c);
    float acc = 0.0f;
#pragma unroll
    for (int j = 0; j < 4; ++j) {
        int tt = t - 3 + j;
        float xv = (tt >= 0) ? qkvz[(size_t)tt * QKVZ_N + col] : 0.0f;
        acc += conv_w[c * 4 + j] * xv;
    }
    y[idx] = siluf_(acc);
}

// ---------------------------------------------------------------------------
// L2-normalize q & k rows in-place in y; q also scaled by DK^-0.5.
// ---------------------------------------------------------------------------
__global__ __launch_bounds__(256) void l2norm_qk_kernel(float* __restrict__ y)
{
    int idx = blockIdx.x * 256 + threadIdx.x;
    if (idx >= TT * 32) return;
    int t = idx >> 5, r = idx & 31;
    float* p = y + (size_t)t * CONVD + ((r < 16) ? (r * DK_) : (KDIM_ + (r - 16) * DK_));
    float ss = 1e-6f;
    for (int d = 0; d < DK_; ++d) ss += p[d] * p[d];
    float inv = rsqrtf(ss);
    if (r < 16) inv *= 0.08838834764831845f;   // 128^-0.5
    for (int d = 0; d < DK_; ++d) p[d] *= inv;
}

// ---------------------------------------------------------------------------
// beta = sigmoid(b), g = -exp(A_log) * softplus(a + dt_bias)
// ---------------------------------------------------------------------------
__global__ __launch_bounds__(256) void prep_gb_kernel(
    const float* __restrict__ ba, const float* __restrict__ dt_bias,
    const float* __restrict__ A_log, float* __restrict__ g, float* __restrict__ beta)
{
    int idx = blockIdx.x * 256 + threadIdx.x;
    if (idx >= TT * HV_) return;
    int t = idx >> 5, h = idx & 31;
    float b = ba[t * 64 + (h >> 1) * 4 + (h & 1)];
    float a = ba[t * 64 + (h >> 1) * 4 + 2 + (h & 1)];
    beta[idx] = sigmoidf_(b);
    float x = a + dt_bias[h];
    float sp = (x > 20.0f) ? x : log1pf(__expf(x));
    g[idx] = -__expf(A_log[h]) * sp;
}

// ---------------------------------------------------------------------------
// Gated delta-rule scan. One block per value head (32 blocks), 256 threads.
// State S[128(K),128(V)] lives in VGPRs.
// ---------------------------------------------------------------------------
__global__ __launch_bounds__(256) void delta_scan_kernel(
    const float* __restrict__ y, const float* __restrict__ g,
    const float* __restrict__ beta, float* __restrict__ core)
{
    const int h   = blockIdx.x;
    const int hk  = h >> 1;
    const int tid = threadIdx.x;
    const int kw  = tid >> 5;
    const int lane = tid & 31;

    __shared__ float sk[128], sq[128], sv[128];
    __shared__ float4 red[8][32];
    __shared__ float sc[2];

    float4 s[16];
#pragma unroll
    for (int i = 0; i < 16; ++i) { s[i].x = 0; s[i].y = 0; s[i].z = 0; s[i].w = 0; }

    for (int t = 0; t < TT; ++t) {
        const float* yt = y + (size_t)t * CONVD;
        if (tid < 128) {
            sk[tid] = yt[KDIM_ + hk * DK_ + tid];
            sv[tid] = yt[2 * KDIM_ + h * DV_ + tid];
        } else {
            int d = tid - 128;
            sq[d] = yt[hk * DK_ + d];
        }
        if (tid == 0) { sc[0] = __expf(g[t * HV_ + h]); sc[1] = beta[t * HV_ + h]; }
        __syncthreads();

        const float eg = sc[0], bt = sc[1];

        float4 p; p.x = p.y = p.z = p.w = 0.0f;
#pragma unroll
        for (int i = 0; i < 16; ++i) {
            float kd = sk[kw * 16 + i];
            s[i].x *= eg; s[i].y *= eg; s[i].z *= eg; s[i].w *= eg;
            p.x += kd * s[i].x; p.y += kd * s[i].y;
            p.z += kd * s[i].z; p.w += kd * s[i].w;
        }
        red[kw][lane] = p;
        __syncthreads();

        float4 kvs; kvs.x = kvs.y = kvs.z = kvs.w = 0.0f;
#pragma unroll
        for (int wi = 0; wi < 8; ++wi) {
            float4 r = red[wi][lane];
            kvs.x += r.x; kvs.y += r.y; kvs.z += r.z; kvs.w += r.w;
        }
        const float4 vt = *(const float4*)&sv[lane * 4];
        float4 delta;
        delta.x = (vt.x - kvs.x) * bt; delta.y = (vt.y - kvs.y) * bt;
        delta.z = (vt.z - kvs.z) * bt; delta.w = (vt.w - kvs.w) * bt;
        __syncthreads();

        float4 po; po.x = po.y = po.z = po.w = 0.0f;
#pragma unroll
        for (int i = 0; i < 16; ++i) {
            float kd = sk[kw * 16 + i];
            float qd = sq[kw * 16 + i];
            s[i].x += kd * delta.x; s[i].y += kd * delta.y;
            s[i].z += kd * delta.z; s[i].w += kd * delta.w;
            po.x += qd * s[i].x; po.y += qd * s[i].y;
            po.z += qd * s[i].z; po.w += qd * s[i].w;
        }
        red[kw][lane] = po;
        __syncthreads();

        if (kw == 0) {
            float4 o; o.x = o.y = o.z = o.w = 0.0f;
#pragma unroll
            for (int wi = 0; wi < 8; ++wi) {
                float4 r = red[wi][lane];
                o.x += r.x; o.y += r.y; o.z += r.z; o.w += r.w;
            }
            *(float4*)&core[(size_t)t * VDIM_ + h * DV_ + lane * 4] = o;
        }
    }
}

// ---------------------------------------------------------------------------
// gated = core * silu(z); RMS-norm over DV; * norm_w.  Block 128 = one row.
// ---------------------------------------------------------------------------
__global__ __launch_bounds__(128) void gate_norm_kernel(
    const float* __restrict__ core, const float* __restrict__ qkvz,
    const float* __restrict__ norm_w, float* __restrict__ gated)
{
    const int row = blockIdx.x;       // t*32 + h
    const int t = row >> 5, h = row & 31;
    const int d = threadIdx.x;
    const float zc = qkvz[(size_t)t * QKVZ_N + (h >> 1) * HEADSTRIDE + 2 * DK_ + 2 * DV_ + (h & 1) * DV_ + d];
    const float gv = core[(size_t)row * DV_ + d] * siluf_(zc);

    __shared__ float ssum[128];
    ssum[d] = gv * gv;
    __syncthreads();
    for (int st = 64; st > 0; st >>= 1) {
        if (d < st) ssum[d] += ssum[d + st];
        __syncthreads();
    }
    const float inv = rsqrtf(ssum[0] * (1.0f / 128.0f) + 1e-6f);
    gated[(size_t)row * DV_ + d] = gv * inv * norm_w[d];
}

// ---------------------------------------------------------------------------
// Host launcher
// ---------------------------------------------------------------------------
extern "C" void kernel_launch(void* const* d_in, const int* in_sizes, int n_in,
                              void* d_out, int out_size, void* d_ws, size_t ws_size,
                              hipStream_t stream)
{
    const float* hidden  = (const float*)d_in[0];   // (1,1024,2048)
    const float* W_qkvz  = (const float*)d_in[1];   // (2048,12288)
    const float* W_ba    = (const float*)d_in[2];   // (2048,64)
    const float* conv_w  = (const float*)d_in[3];   // (8192,4)
    const float* dt_bias = (const float*)d_in[4];   // (32,)
    const float* A_log   = (const float*)d_in[5];   // (32,)
    const float* norm_w  = (const float*)d_in[6];   // (128,)
    const float* W_out   = (const float*)d_in[7];   // (4096,2048)
    float* out = (float*)d_out;                     // (1,1024,2048)

    size_t off = 0;
    auto carve = [&](size_t bytes) -> char* {
        char* p = (char*)d_ws + off;
        off = (off + bytes + 255) & ~(size_t)255;
        return p;
    };
    float*  ws_qkvz   = (float*)carve((size_t)TT * QKVZ_N * sizeof(float));    // 50 MB
    float*  ws_ba     = (float*)carve((size_t)TT * 64 * sizeof(float));
    float*  ws_y      = (float*)carve((size_t)TT * CONVD * sizeof(float));     // 32 MB
    float*  ws_g      = (float*)carve((size_t)TT * HV_ * sizeof(float));
    float*  ws_beta   = (float*)carve((size_t)TT * HV_ * sizeof(float));
    float*  ws_core   = (float*)carve((size_t)TT * VDIM_ * sizeof(float));     // 16 MB
    float*  ws_gated  = (float*)carve((size_t)TT * VDIM_ * sizeof(float));     // 16 MB
    __bf16* ws_hbf    = (__bf16*)carve((size_t)TT * HDIM * sizeof(__bf16));    //  4 MB
    __bf16* ws_gbf    = (__bf16*)carve((size_t)TT * VDIM_ * sizeof(__bf16));   //  8 MB
    __bf16* ws_WqkvzT = (__bf16*)carve((size_t)QKVZ_N * HDIM * sizeof(__bf16));// 48 MB
    __bf16* ws_WbaT   = (__bf16*)carve((size_t)64 * HDIM * sizeof(__bf16));
    __bf16* ws_WoutT  = (__bf16*)carve((size_t)HDIM * VDIM_ * sizeof(__bf16)); // 16 MB
    (void)ws_size; (void)in_sizes; (void)n_in; (void)out_size;

    // 0) precision prep: A operands to bf16 row-major, weights to bf16 [N,K]
    cvt_bf16_kernel<<<(TT * HDIM + 255) / 256, 256, 0, stream>>>(hidden, ws_hbf, TT * HDIM);
    transpose_cvt_bf16_kernel<<<dim3(QKVZ_N / 32, HDIM / 32), 256, 0, stream>>>(W_qkvz, ws_WqkvzT, HDIM, QKVZ_N);
    transpose_cvt_bf16_kernel<<<dim3(64 / 32,    HDIM / 32), 256, 0, stream>>>(W_ba,   ws_WbaT,   HDIM, 64);
    transpose_cvt_bf16_kernel<<<dim3(HDIM / 32, VDIM_ / 32), 256, 0, stream>>>(W_out,  ws_WoutT,  VDIM_, HDIM);

    // 1) qkvz = hidden @ W_qkvz   (1024x2048x12288)
    wmma_gemm_bf16<<<dim3(QKVZ_N / 64, TT / 128), 256, 0, stream>>>(
        ws_hbf, ws_WqkvzT, ws_qkvz, TT, QKVZ_N, HDIM);

    // 2) ba = hidden @ W_ba       (1024x2048x64)
    wmma_gemm_bf16<<<dim3(1, TT / 128), 256, 0, stream>>>(
        ws_hbf, ws_WbaT, ws_ba, TT, 64, HDIM);

    // 3) depthwise causal conv + SiLU  -> y (1024 x 8192)
    conv_silu_kernel<<<(TT * CONVD + 255) / 256, 256, 0, stream>>>(ws_qkvz, conv_w, ws_y);

    // 4) l2norm q,k (in place in y)
    l2norm_qk_kernel<<<(TT * 32 + 255) / 256, 256, 0, stream>>>(ws_y);

    // 5) beta / g
    prep_gb_kernel<<<(TT * HV_ + 255) / 256, 256, 0, stream>>>(ws_ba, dt_bias, A_log, ws_g, ws_beta);

    // 6) recurrent delta-rule scan  -> core (1024 x 4096)
    delta_scan_kernel<<<HV_, 256, 0, stream>>>(ws_y, ws_g, ws_beta, ws_core);

    // 7) silu(z) gate + RMS norm -> gated
    gate_norm_kernel<<<TT * HV_, 128, 0, stream>>>(ws_core, ws_qkvz, norm_w, ws_gated);

    // 8) out = gated @ W_out      (1024x4096x2048)
    cvt_bf16_kernel<<<(TT * VDIM_ + 255) / 256, 256, 0, stream>>>(ws_gated, ws_gbf, TT * VDIM_);
    wmma_gemm_bf16<<<dim3(HDIM / 64, TT / 128), 256, 0, stream>>>(
        ws_gbf, ws_WoutT, out, TT, HDIM, VDIM_);
}
